// GRUCell_19722489823240
// MI455X (gfx1250) — compile-verified, matrix operand order checked by reading, not run on hardware
//
#include <hip/hip_runtime.h>
#include <hip/hip_bf16.h>

typedef __attribute__((ext_vector_type(16))) _Float16 v16h;
typedef __attribute__((ext_vector_type(8)))  float    v8f;

#define BATCH 65536
#define IDIM  64
#define HDIM  128

// packed-fragment counts: frag = 32 lanes x 16 halves = 512 halves = 1KB
#define WX_KSTEPS 2
#define WH_KSTEPS 4
#define W3_KSTEPS 4
#define WX_FRAGS  (24 * WX_KSTEPS)   // 384/16 n-tiles
#define WH_FRAGS  (16 * WH_KSTEPS)   // 256/16 n-tiles
#define W3_FRAGS  (8  * W3_KSTEPS)   // 128/16 n-tiles

// ---------------- quantization helpers (match jax reference) ----------------
__device__ __forceinline__ void qparams(float m, float& sc, float& inv, float& thr) {
    float dv = 0.0f;
    if (m > 0.0f) { int e = 0; (void)frexpf(m, &e); dv = (float)e; } // floor(log2 m)+1
    float q = fmaxf(4.0f - dv, 0.0f);
    sc  = exp2f(q);
    inv = exp2f(-q);
    thr = exp2f(4.0f - q) - inv;
}
__device__ __forceinline__ float dynq(float x, float sc, float inv, float thr) {
    float o = truncf(x * sc) * inv;
    return fminf(fmaxf(o, -thr), thr);
}
__device__ __forceinline__ float sigq(float x, float sc, float inv, float thr) {
    float qx = dynq(x, sc, inv, thr);
    float o  = rintf(qx * 3.0f) * (1.0f / 15.0f) + 0.5f;   // round(qx*15*0.2)/15 + 0.5
    return fminf(fmaxf(o, 0.0f), 1.0f);
}
__device__ __forceinline__ float tanhq(float x) {
    float o = rintf(x * 15.0f) * (1.0f / 15.0f);
    return fminf(fmaxf(o, -1.0f), 1.0f);
}

__device__ __forceinline__ float waveMax(float v) {
#pragma unroll
    for (int ofs = 16; ofs > 0; ofs >>= 1)
        v = fmaxf(v, __shfl_xor(v, ofs, 32));
    return v;
}
// all reduced values are >= 0: uint-bit compare == float compare
__device__ __forceinline__ void atomicMaxPos(float* p, float v) {
    atomicMax(reinterpret_cast<unsigned int*>(p), __float_as_uint(v));
}

// ---------------- WMMA fragment helpers (ISA 7.12.2 layouts) ----------------
// A 16x32 f16: lane L(<16) holds row M=L, K in {kb..kb+7} U {kb+16..kb+23};
//              lane L+16 holds row M=L, K in {kb+8..kb+15} U {kb+24..kb+31}.
__device__ __forceinline__ v16h load_a_f32(const float* base, int ld, int row0, int kb) {
    int lane = threadIdx.x & 31;
    int m    = lane & 15;
    int ko   = (lane & 16) ? 8 : 0;
    const float* p = base + (size_t)(row0 + m) * ld + kb + ko;
    v16h a; _Float16* ah = reinterpret_cast<_Float16*>(&a);
#pragma unroll
    for (int e = 0; e < 8; ++e) { ah[e] = (_Float16)p[e]; ah[8 + e] = (_Float16)p[16 + e]; }
    return a;
}
__device__ __forceinline__ v16h load_a_quant(const float* base, int ld, int row0, int kb,
                                             float sc, float inv, float thr) {
    int lane = threadIdx.x & 31;
    int m    = lane & 15;
    int ko   = (lane & 16) ? 8 : 0;
    const float* p = base + (size_t)(row0 + m) * ld + kb + ko;
    v16h a; _Float16* ah = reinterpret_cast<_Float16*>(&a);
#pragma unroll
    for (int e = 0; e < 8; ++e) {
        ah[e]     = (_Float16)dynq(p[e],      sc, inv, thr);
        ah[8 + e] = (_Float16)dynq(p[16 + e], sc, inv, thr);
    }
    return a;
}
// B fragments pre-packed in register image: [frag][lane][16 halves]; a single
// 32-byte per-lane load (2x global_load_b128), zero conversion VALU.
__device__ __forceinline__ v16h load_b_packed(const _Float16* __restrict__ base, int frag) {
    int lane = threadIdx.x & 31;
    return *reinterpret_cast<const v16h*>(base + ((size_t)frag << 9) + (lane << 4));
}
__device__ __forceinline__ v8f wmma(v16h a, v16h b, v8f c) {
    return __builtin_amdgcn_wmma_f32_16x16x32_f16(false, a, false, b, (short)0, c, false, false);
}

// ---------------- kernel: pack one weight fragment per wave ----------------
// B 32x16: lane n holds column N=n, same K split as A; since gate = act @ W.T,
// B[k,n] = W[nb+n, k]. fragIdx = ntile*ksteps + kstep.
__global__ void k_pack_w(const float* __restrict__ W, _Float16* __restrict__ out,
                         int ld, int ksteps) {
    int frag = blockIdx.x;
    int nt = frag / ksteps, kt = frag - nt * ksteps;
    int lane = threadIdx.x & 31;
    int n  = lane & 15;
    int ko = (lane & 16) ? 8 : 0;
    const float* p = W + (size_t)(nt * 16 + n) * ld + kt * 32 + ko;
    v16h f; _Float16* fh = reinterpret_cast<_Float16*>(&f);
#pragma unroll
    for (int e = 0; e < 8; ++e) { fh[e] = (_Float16)p[e]; fh[8 + e] = (_Float16)p[16 + e]; }
    *reinterpret_cast<v16h*>(out + ((size_t)frag << 9) + (lane << 4)) = f;
}

// ---------------- kernel: zero the 4 ws scalars ----------------
__global__ void k_init(float* ws) {
    if (threadIdx.x < 4) ws[threadIdx.x] = 0.0f;
}

// ---------------- kernel: ws[0] = max|x| ----------------
__global__ void k_maxabs_x(const float* __restrict__ x, int n4, float* ws) {
    const float4* x4 = reinterpret_cast<const float4*>(x);
    float m = 0.0f;
    for (int i = blockIdx.x * blockDim.x + threadIdx.x; i < n4; i += gridDim.x * blockDim.x) {
        float4 v = x4[i];
        m = fmaxf(m, fmaxf(fmaxf(fabsf(v.x), fabsf(v.y)), fmaxf(fabsf(v.z), fabsf(v.w))));
    }
    m = waveMax(m);
    if ((threadIdx.x & 31) == 0) atomicMaxPos(&ws[0], m);
}

// ---------------- kernel: ws[1]=max|i_r+h_r|, ws[2]=max|i_i+h_i| ----------
__global__ void __launch_bounds__(256) k_gate_max(
        const float* __restrict__ x, const float* __restrict__ hidden,
        const _Float16* __restrict__ wxp, const float* __restrict__ bx,
        const _Float16* __restrict__ whp, const float* __restrict__ bh,
        float* ws) {
    int wave = threadIdx.x >> 5;
    int row  = (blockIdx.x * 8 + wave) * 16;
    int n    = threadIdx.x & 15;

    __builtin_prefetch(wxp, 0, 0);   // packed weights: L2/WGP$ resident
    __builtin_prefetch(whp, 0, 0);

    float scx, invx, thrx; qparams(ws[0], scx, invx, thrx);
    v16h ax0 = load_a_quant(x, IDIM, row, 0,  scx, invx, thrx);
    v16h ax1 = load_a_quant(x, IDIM, row, 32, scx, invx, thrx);
    v16h ah[4];
#pragma unroll
    for (int k = 0; k < 4; ++k) ah[k] = load_a_f32(hidden, HDIM, row, k * 32);

    float mr = 0.0f, mi = 0.0f;
#pragma unroll
    for (int t = 0; t < 8; ++t) {
        v8f ar = {};
        ar = wmma(ax0, load_b_packed(wxp, t * WX_KSTEPS + 0), ar);
        ar = wmma(ax1, load_b_packed(wxp, t * WX_KSTEPS + 1), ar);
#pragma unroll
        for (int k = 0; k < 4; ++k)
            ar = wmma(ah[k], load_b_packed(whp, t * WH_KSTEPS + k), ar);
        float br = bx[t * 16 + n] + bh[t * 16 + n];
#pragma unroll
        for (int v = 0; v < 8; ++v) mr = fmaxf(mr, fabsf(ar[v] + br));

        v8f ai = {};
        ai = wmma(ax0, load_b_packed(wxp, (8 + t) * WX_KSTEPS + 0), ai);
        ai = wmma(ax1, load_b_packed(wxp, (8 + t) * WX_KSTEPS + 1), ai);
#pragma unroll
        for (int k = 0; k < 4; ++k)
            ai = wmma(ah[k], load_b_packed(whp, (8 + t) * WH_KSTEPS + k), ai);
        float bi = bx[HDIM + t * 16 + n] + bh[HDIM + t * 16 + n];
#pragma unroll
        for (int v = 0; v < 8; ++v) mi = fmaxf(mi, fabsf(ai[v] + bi));
    }
    mr = waveMax(mr); mi = waveMax(mi);
    if ((threadIdx.x & 31) == 0) { atomicMaxPos(&ws[1], mr); atomicMaxPos(&ws[2], mi); }
}

// ---------------- kernel: fused gates -> rh -> GEMM3 -> hy ----------------
#define CW 4
__global__ void __launch_bounds__(128) k_fused(
        const float* __restrict__ x, const float* __restrict__ hidden,
        const _Float16* __restrict__ wxp, const float* __restrict__ bx,
        const _Float16* __restrict__ whp, const float* __restrict__ bh,
        const _Float16* __restrict__ w3p, const float* __restrict__ b3,
        float* __restrict__ out, float* ws) {
    __shared__ _Float16 s_rh[CW][16 * HDIM];
    __shared__ _Float16 s_ig[CW][16 * HDIM];
    __shared__ _Float16 s_in[CW][16 * HDIM];

    int wave = threadIdx.x >> 5;
    int lane = threadIdx.x & 31;
    int row  = (blockIdx.x * CW + wave) * 16;
    int n    = lane & 15;
    int mofs = (lane & 16) ? 8 : 0;

    float scx, invx, thrx; qparams(ws[0], scx, invx, thrx);
    float scr, invr, thrr; qparams(ws[1], scr, invr, thrr);
    float sci, invi, thri; qparams(ws[2], sci, invi, thri);

    v16h ax0 = load_a_quant(x, IDIM, row, 0,  scx, invx, thrx);
    v16h ax1 = load_a_quant(x, IDIM, row, 32, scx, invx, thrx);
    v16h ah[4];
#pragma unroll
    for (int k = 0; k < 4; ++k) ah[k] = load_a_f32(hidden, HDIM, row, k * 32);

    _Float16* rh = s_rh[wave];
    _Float16* ig = s_ig[wave];
    _Float16* in = s_in[wave];

#pragma unroll
    for (int t = 0; t < 8; ++t) {
        int cb = t * 16;
        v8f ar = {}, ai = {}, an = {};
        ar = wmma(ax0, load_b_packed(wxp, t * WX_KSTEPS + 0),        ar);
        ar = wmma(ax1, load_b_packed(wxp, t * WX_KSTEPS + 1),        ar);
        ai = wmma(ax0, load_b_packed(wxp, (8 + t) * WX_KSTEPS + 0),  ai);
        ai = wmma(ax1, load_b_packed(wxp, (8 + t) * WX_KSTEPS + 1),  ai);
        an = wmma(ax0, load_b_packed(wxp, (16 + t) * WX_KSTEPS + 0), an);
        an = wmma(ax1, load_b_packed(wxp, (16 + t) * WX_KSTEPS + 1), an);
#pragma unroll
        for (int k = 0; k < 4; ++k) {
            ar = wmma(ah[k], load_b_packed(whp, t * WH_KSTEPS + k),       ar);
            ai = wmma(ah[k], load_b_packed(whp, (8 + t) * WH_KSTEPS + k), ai);
        }
        float br = bx[cb + n] + bh[cb + n];
        float bi = bx[HDIM + cb + n] + bh[HDIM + cb + n];
        float bn = bx[2 * HDIM + cb + n];
#pragma unroll
        for (int v = 0; v < 8; ++v) {
            int r = v + mofs;                                   // C-layout: (M=r, N=cb+n)
            float hid = hidden[(size_t)(row + r) * HDIM + cb + n];
            float rg  = sigq(ar[v] + br, scr, invr, thrr);
            float igv = sigq(ai[v] + bi, sci, invi, thri);
            rh[r * HDIM + cb + n] = (_Float16)(rg * hid);
            ig[r * HDIM + cb + n] = (_Float16)igv;
            in[r * HDIM + cb + n] = (_Float16)(an[v] + bn);
        }
    }
    __syncthreads();

    // rh: C-layout -> A-fragment layout (the LDS transpose)
    v16h arh[4];
    {
        int m  = lane & 15;
        int ko = (lane & 16) ? 8 : 0;
#pragma unroll
        for (int k = 0; k < 4; ++k) {
            _Float16* fh = reinterpret_cast<_Float16*>(&arh[k]);
            const _Float16* p = rh + m * HDIM + k * 32 + ko;
#pragma unroll
            for (int e = 0; e < 8; ++e) { fh[e] = p[e]; fh[8 + e] = p[16 + e]; }
        }
    }

    float mh = 0.0f;
#pragma unroll
    for (int t = 0; t < 8; ++t) {
        int cb = t * 16;
        v8f ac = {};
#pragma unroll
        for (int k = 0; k < 4; ++k)
            ac = wmma(arh[k], load_b_packed(w3p, t * W3_KSTEPS + k), ac);
        float bc = b3[cb + n];
#pragma unroll
        for (int v = 0; v < 8; ++v) {
            int r = v + mofs;
            float cand = ac[v] + bc;
            float ing  = (float)in[r * HDIM + cb + n];
            float newg = tanhq(ing + cand);
            float hid  = hidden[(size_t)(row + r) * HDIM + cb + n];
            float igv  = (float)ig[r * HDIM + cb + n];
            float hy   = newg + igv * (hid - newg);
            out[(size_t)(row + r) * HDIM + cb + n] = hy;
            mh = fmaxf(mh, fabsf(hy));
        }
    }
    mh = waveMax(mh);
    if (lane == 0) atomicMaxPos(&ws[3], mh);
}

// ---------------- kernel: out = act_quant(hy) in place ----------------
__global__ void k_quant_out(float* __restrict__ out, int n4, const float* ws) {
    float sc, inv, thr; qparams(ws[3], sc, inv, thr);
    float4* o4 = reinterpret_cast<float4*>(out);
    int i = blockIdx.x * blockDim.x + threadIdx.x;
    if (i < n4) {
        float4 v = o4[i];
        v.x = dynq(v.x, sc, inv, thr);
        v.y = dynq(v.y, sc, inv, thr);
        v.z = dynq(v.z, sc, inv, thr);
        v.w = dynq(v.w, sc, inv, thr);
        o4[i] = v;
    }
}

extern "C" void kernel_launch(void* const* d_in, const int* in_sizes, int n_in,
                              void* d_out, int out_size, void* d_ws, size_t ws_size,
                              hipStream_t stream) {
    const float* x      = (const float*)d_in[0];
    const float* hidden = (const float*)d_in[1];
    const float* wx     = (const float*)d_in[2];
    const float* bx     = (const float*)d_in[3];
    const float* wh     = (const float*)d_in[4];
    const float* bh     = (const float*)d_in[5];
    const float* w3     = (const float*)d_in[6];
    const float* b3     = (const float*)d_in[7];
    float* out = (float*)d_out;
    float* ws  = (float*)d_ws;

    // ws layout: [0,16) 4 scalars; packed f16 weights from byte 64 (aligned).
    _Float16* wxp = reinterpret_cast<_Float16*>((char*)d_ws + 64);
    _Float16* whp = wxp + (size_t)WX_FRAGS * 512;
    _Float16* w3p = whp + (size_t)WH_FRAGS * 512;

    k_init<<<1, 32, 0, stream>>>(ws);
    k_pack_w<<<WX_FRAGS, 32, 0, stream>>>(wx, wxp, IDIM, WX_KSTEPS);
    k_pack_w<<<WH_FRAGS, 32, 0, stream>>>(wh, whp, HDIM, WH_KSTEPS);
    k_pack_w<<<W3_FRAGS, 32, 0, stream>>>(w3, w3p, HDIM, W3_KSTEPS);
    k_maxabs_x<<<2048, 256, 0, stream>>>(x, BATCH * IDIM / 4, ws);
    k_gate_max<<<BATCH / (16 * 8), 256, 0, stream>>>(x, hidden, wxp, bx, whp, bh, ws);
    k_fused<<<BATCH / (16 * CW), 128, 0, stream>>>(x, hidden, wxp, bx, whp, bh, w3p, b3, out, ws);
    k_quant_out<<<(BATCH * HDIM / 4) / 256, 256, 0, stream>>>(out, BATCH * HDIM / 4, ws);
}
